// GraphEncoderTL_87625922773142
// MI455X (gfx1250) — compile-verified, matrix operand order checked by reading, not run on hardware
//
#include <hip/hip_runtime.h>

// Dead-code-eliminated GraphEncoderTL: out = adj @ (raw_feat @ W_gat) + b_gat
// N=16000, D=128.
// Big GEMM: V_WMMA_F32_16X16X32_BF16 with 2-word bf16 split
// (Ah*Bh + Ah*Bl + Al*Bh, fp32 accum) => ~fp32 accuracy at bf16 matrix rate
// (64 FLOP/byte intensity vs 23.3 TB/s HBM => need bf16-class matrix rate).
// X slabs double-buffered in LDS via GLOBAL_LOAD_ASYNC_TO_LDS_B128 (ASYNCcnt),
// overlapping the async fill of chunk c+1 with compute of chunk c.

#define NN 16000
#define DD 128
#define KT 64            // K-rows of X staged in LDS per chunk
#define BT 128           // threads per block in adj_gemm (4 waves, 64 rows)

typedef __attribute__((ext_vector_type(2)))  float  v2f;
typedef __attribute__((ext_vector_type(8)))  float  v8f;
typedef __attribute__((ext_vector_type(4)))  int    v4i;
typedef __attribute__((ext_vector_type(16))) __bf16 v16bf;
typedef __attribute__((ext_vector_type(2)))  __bf16 v2bf;

// global (AS1) / LDS (AS3) qualified v4i pointers for the async-to-LDS builtin
typedef v4i __attribute__((address_space(1)))* gv4i_p;
typedef v4i __attribute__((address_space(3)))* lv4i_p;

#if __has_builtin(__builtin_amdgcn_global_load_async_to_lds_b128)
#define HAVE_ASYNC_LDS 1
#else
#define HAVE_ASYNC_LDS 0
#endif

__device__ __forceinline__ void wait_asynccnt0() {
#if __has_builtin(__builtin_amdgcn_s_wait_asynccnt)
  __builtin_amdgcn_s_wait_asynccnt(0);
#else
  asm volatile("s_wait_asynccnt 0x0" ::: "memory");
#endif
}

__device__ __forceinline__ v8f wmma4(v2f a, v2f b, v8f c) {
  return __builtin_amdgcn_wmma_f32_16x16x4_f32(false, a, false, b, (short)0, c,
                                               false, false);
}

__device__ __forceinline__ v8f wmma_bf16(v16bf a, v16bf b, v8f c) {
  return __builtin_amdgcn_wmma_f32_16x16x32_bf16(false, a, false, b, (short)0,
                                                 c, false, false);
}

// ---------------------------------------------------------------------------
// Kernel 1: raw_feat = concat(lane_tab[lane], type_tab[type],
//                             length_tab[len], node_tab[node])  -> [N, 128]
// ---------------------------------------------------------------------------
__global__ void __launch_bounds__(256)
gather_concat_kernel(const int* __restrict__ node_f, const int* __restrict__ type_f,
                     const int* __restrict__ len_f,  const int* __restrict__ lane_f,
                     const float* __restrict__ node_tab, const float* __restrict__ type_tab,
                     const float* __restrict__ len_tab,  const float* __restrict__ lane_tab,
                     float* __restrict__ raw) {
  int i = blockIdx.x * 256 + threadIdx.x;
  int node = i >> 5;
  if (node >= NN) return;
  int c = (i & 31) << 2;
  const float* src;
  int sub;
  if (c < 32)      { src = lane_tab + (size_t)lane_f[node] * 32; sub = c; }
  else if (c < 64) { src = type_tab + (size_t)type_f[node] * 32; sub = c - 32; }
  else if (c < 96) { src = len_tab  + (size_t)len_f[node]  * 32; sub = c - 64; }
  else             { src = node_tab + (size_t)node_f[node] * 32; sub = c - 96; }
  const float4 v = *reinterpret_cast<const float4*>(src + sub);
  *reinterpret_cast<float4*>(raw + (size_t)node * DD + c) = v;
}

// ---------------------------------------------------------------------------
// Kernel 2: X = raw_feat @ W_gat  [16000,128]x[128,128]  (exact fp32 WMMA)
// ---------------------------------------------------------------------------
__global__ void __launch_bounds__(256)
feat_gemm_kernel(const float* __restrict__ raw, const float* __restrict__ W,
                 float* __restrict__ X) {
  const int wave = threadIdx.x >> 5;
  const int lane = threadIdx.x & 31;
  const int m    = lane & 15;
  const int hi   = lane >> 4;
  const int kofs = hi * 2;
  const int row0 = blockIdx.x * 128 + wave * 16;

  v8f acc[8] = {};
  const float* arow = raw + (size_t)(row0 + m) * DD + kofs;
#pragma unroll 4
  for (int k = 0; k < DD; k += 4) {
    v2f a = *reinterpret_cast<const v2f*>(arow + k);
#pragma unroll
    for (int n = 0; n < 8; ++n) {
      const int col = n * 16 + m;
      v2f b;
      b.x = W[(k + kofs)     * DD + col];
      b.y = W[(k + kofs + 1) * DD + col];
      acc[n] = wmma4(a, b, acc[n]);
    }
  }
#pragma unroll
  for (int n = 0; n < 8; ++n) {
    const int col = n * 16 + m;
#pragma unroll
    for (int r = 0; r < 8; ++r)
      X[(size_t)(row0 + r + 8 * hi) * DD + col] = acc[n][r];
  }
}

// ---------------------------------------------------------------------------
// Kernel 3: split X into (hi, lo) bf16 and pre-swizzle into B-fragment-major
// layout. Element (k, c):
//   t=k>>5, kk=k&31, n=c>>4, nn=c&15, khalf=(kk>>3)&1,
//   v=(kk&16 ? 4:0)+((kk&7)>>1), p=kk&1, lane=nn+16*khalf
//   off = ((t*8+n)*32+lane)*16 + v*2 + p      (bf16 units)
// => each lane's B-fragment (16 bf16 = 32 B) is contiguous.
// ---------------------------------------------------------------------------
__global__ void __launch_bounds__(256)
xsplit_swizzle_kernel(const float* __restrict__ X, __bf16* __restrict__ xh,
                      __bf16* __restrict__ xl) {
  int i = blockIdx.x * 256 + threadIdx.x;
  if (i >= (NN / 2) * DD) return;
  int c = i % DD;
  int k = (i / DD) * 2;
  float x0 = X[(size_t)k * DD + c];
  float x1 = X[(size_t)(k + 1) * DD + c];
  __bf16 h0 = (__bf16)x0, h1 = (__bf16)x1;
  __bf16 l0 = (__bf16)(x0 - (float)h0), l1 = (__bf16)(x1 - (float)h1);

  int t = k >> 5, kk = k & 31;
  int n = c >> 4, nn = c & 15;
  int khalf = (kk >> 3) & 1;
  int v = ((kk & 16) >> 2) + ((kk & 7) >> 1);
  int lane = nn + 16 * khalf;
  size_t off = ((size_t)(t * 8 + n) * 32 + lane) * 16 + v * 2;

  v2bf ph = {h0, h1};
  v2bf pl = {l0, l1};
  *reinterpret_cast<v2bf*>(xh + off) = ph;
  *reinterpret_cast<v2bf*>(xl + off) = pl;
}

// ---------------------------------------------------------------------------
// Kernel 4: out = adj @ X + b_gat  via bf16-split WMMA.
// 4 waves/block, 16 rows/wave -> 64 rows/block, 250 blocks.
// Double-buffered LDS slabs (2 x 32 KB): async fill of chunk c+1 overlaps
// compute of chunk c; one barrier per chunk. adj split hi/lo on the fly.
// ---------------------------------------------------------------------------
__global__ void __launch_bounds__(BT)
adj_gemm_kernel(const float* __restrict__ adj,
                const __bf16* __restrict__ xh, const __bf16* __restrict__ xl,
                const float* __restrict__ bias, float* __restrict__ out) {
  __shared__ __bf16 sh[2][KT * DD];   // 2 x 16 KB swizzled hi slabs
  __shared__ __bf16 sl[2][KT * DD];   // 2 x 16 KB swizzled lo slabs

  const int wave = threadIdx.x >> 5;
  const int lane = threadIdx.x & 31;
  const int m    = lane & 15;
  const int hi   = lane >> 4;
  const int row0 = blockIdx.x * 64 + wave * 16;

  v8f acc[8] = {};

  // A-fragment base: lane needs K = kbase + 8*hi + {0..7} (elems 0-7)
  //                  and K = kbase + 16 + 8*hi + {0..7}  (elems 8-15)
  const float* arow = adj + (size_t)(row0 + m) * NN + 8 * hi;

  // stage swizzled X slabs for K-chunk starting at k0 into buffer `buf`
  auto issue_slab = [&](int buf, int k0) {
#if HAVE_ASYNC_LDS
    gv4i_p gh = (gv4i_p)(xh + (size_t)k0 * DD);
    gv4i_p gl = (gv4i_p)(xl + (size_t)k0 * DD);
    lv4i_p lh = (lv4i_p)(sh[buf]);
    lv4i_p ll = (lv4i_p)(sl[buf]);
#pragma unroll
    for (int i = 0; i < (KT * DD * 2 / 16) / BT; ++i) {
      int o = threadIdx.x + i * BT;       // v4i (16 B) units
      __builtin_amdgcn_global_load_async_to_lds_b128(gh + o, lh + o, 0, 0);
      __builtin_amdgcn_global_load_async_to_lds_b128(gl + o, ll + o, 0, 0);
    }
#else
    const uint4* srch = reinterpret_cast<const uint4*>(xh + (size_t)k0 * DD);
    const uint4* srcl = reinterpret_cast<const uint4*>(xl + (size_t)k0 * DD);
    uint4* dsth = reinterpret_cast<uint4*>(sh[buf]);
    uint4* dstl = reinterpret_cast<uint4*>(sl[buf]);
#pragma unroll
    for (int i = 0; i < (KT * DD * 2 / 16) / BT; ++i) {
      int o = threadIdx.x + i * BT;
      dsth[o] = srch[o];
      dstl[o] = srcl[o];
    }
#endif
  };

  const int nchunks = NN / KT;          // 250
  issue_slab(0, 0);

  for (int c = 0; c < nchunks; ++c) {
    const int p  = c & 1;
    const int k0 = c * KT;

#if HAVE_ASYNC_LDS
    wait_asynccnt0();                   // own async loads for slab p complete
#endif
    __syncthreads();                    // all waves' slab-p data visible;
                                        // all waves done reading slab p^1
    if (c + 1 < nchunks)
      issue_slab(p ^ 1, k0 + KT);       // overlap fill of c+1 with compute of c

    if (k0 + KT < NN)
      __builtin_prefetch(arow + k0 + KT, 0, 0);

#pragma unroll
    for (int t = 0; t < KT / 32; ++t) {
      const int kbase = k0 + t * 32;
      // load 16 fp32 of adj for this lane's A-fragment
      float va[16];
      *reinterpret_cast<float4*>(va + 0)  = *reinterpret_cast<const float4*>(arow + kbase + 0);
      *reinterpret_cast<float4*>(va + 4)  = *reinterpret_cast<const float4*>(arow + kbase + 4);
      *reinterpret_cast<float4*>(va + 8)  = *reinterpret_cast<const float4*>(arow + kbase + 16);
      *reinterpret_cast<float4*>(va + 12) = *reinterpret_cast<const float4*>(arow + kbase + 20);
      // split into hi/lo bf16 fragments
      v16bf ah, al;
#pragma unroll
      for (int j = 0; j < 16; ++j) {
        float  x  = va[j];
        __bf16 hb = (__bf16)x;
        ah[j] = hb;
        al[j] = (__bf16)(x - (float)hb);
      }
      const __bf16* bbase_h = sh[p] + ((size_t)t * 8 * 32 + lane) * 16;
      const __bf16* bbase_l = sl[p] + ((size_t)t * 8 * 32 + lane) * 16;
      // interleave column-tile pairs so consecutive WMMAs hit different
      // accumulators (no back-to-back C->C dependency on the XDL pipe)
#pragma unroll
      for (int n = 0; n < 8; n += 2) {
        v16bf bh0 = *reinterpret_cast<const v16bf*>(bbase_h + (size_t)(n + 0) * 32 * 16);
        v16bf bh1 = *reinterpret_cast<const v16bf*>(bbase_h + (size_t)(n + 1) * 32 * 16);
        v16bf bl0 = *reinterpret_cast<const v16bf*>(bbase_l + (size_t)(n + 0) * 32 * 16);
        v16bf bl1 = *reinterpret_cast<const v16bf*>(bbase_l + (size_t)(n + 1) * 32 * 16);
        acc[n + 0] = wmma_bf16(ah, bh0, acc[n + 0]);   // Ah*Bh
        acc[n + 1] = wmma_bf16(ah, bh1, acc[n + 1]);
        acc[n + 0] = wmma_bf16(ah, bl0, acc[n + 0]);   // Ah*Bl
        acc[n + 1] = wmma_bf16(ah, bl1, acc[n + 1]);
        acc[n + 0] = wmma_bf16(al, bh0, acc[n + 0]);   // Al*Bh
        acc[n + 1] = wmma_bf16(al, bh1, acc[n + 1]);
      }
    }
  }

#pragma unroll
  for (int n = 0; n < 8; ++n) {
    const int col = n * 16 + m;
    const float bv = bias[col];
#pragma unroll
    for (int r = 0; r < 8; ++r)
      out[(size_t)(row0 + r + 8 * hi) * DD + col] = acc[n][r] + bv;
  }
}

// ---------------------------------------------------------------------------
extern "C" void kernel_launch(void* const* d_in, const int* in_sizes, int n_in,
                              void* d_out, int out_size, void* d_ws, size_t ws_size,
                              hipStream_t stream) {
  const int*   node_f   = (const int*)d_in[0];
  const int*   type_f   = (const int*)d_in[1];
  const int*   len_f    = (const int*)d_in[2];
  const int*   lane_f   = (const int*)d_in[3];
  const float* adj      = (const float*)d_in[4];
  const float* node_tab = (const float*)d_in[8];
  const float* type_tab = (const float*)d_in[9];
  const float* len_tab  = (const float*)d_in[10];
  const float* lane_tab = (const float*)d_in[11];
  const float* W_gat    = (const float*)d_in[16];
  const float* b_gat    = (const float*)d_in[17];

  float* out = (float*)d_out;
  float*  raw = (float*)d_ws;                          // [N,128] fp32 (8 MB)
  float*  X   = raw + (size_t)NN * DD;                 // [N,128] fp32 (8 MB)
  __bf16* xh  = (__bf16*)(X + (size_t)NN * DD);        // [N,128] bf16 (4 MB)
  __bf16* xl  = xh + (size_t)NN * DD;                  // [N,128] bf16 (4 MB)

  gather_concat_kernel<<<(NN * 32 + 255) / 256, 256, 0, stream>>>(
      node_f, type_f, len_f, lane_f, node_tab, type_tab, len_tab, lane_tab, raw);

  feat_gemm_kernel<<<NN / 128, 256, 0, stream>>>(raw, W_gat, X);

  xsplit_swizzle_kernel<<<((NN / 2) * DD + 255) / 256, 256, 0, stream>>>(X, xh, xl);

  adj_gemm_kernel<<<NN / 64, BT, 0, stream>>>(adj, xh, xl, b_gat, out);
}
